// Encoder_61521111548392
// MI455X (gfx1250) — matrix operand order repaired and checked by reading, not compiled, not run on hardware
//
#include <hip/hip_runtime.h>

typedef __attribute__((ext_vector_type(16))) _Float16 v16h;
typedef __attribute__((ext_vector_type(8)))  _Float16 v8h;
typedef __attribute__((ext_vector_type(8)))  float    v8f;

#define N_NODES 10000
#define N_EDGES 320000
#define HID     128

__device__ __forceinline__ float lrelu(float v) { return v >= 0.f ? v : 0.1f * v; }

__device__ __forceinline__ v16h packA(v8h lo, v8h hi) {
  v16h r;
#pragma unroll
  for (int i = 0; i < 8; i++) { r[i] = lo[i]; r[i + 8] = hi[i]; }
  return r;
}

// dst[n*Kpad + k] = (k<K) ? f16(src[k*N + n]) : 0   (store W^T in f16, K padded)
__global__ void k_transpose_f16(_Float16* __restrict__ dst, const float* __restrict__ src,
                                int K, int Nc, int Kpad) {
  int idx = blockIdx.x * blockDim.x + threadIdx.x;
  int total = Nc * Kpad;
  if (idx >= total) return;
  int n = idx / Kpad, k = idx % Kpad;
  dst[(size_t)n * Kpad + k] = (k < K) ? (_Float16)src[(size_t)k * Nc + n] : (_Float16)0.f;
}

__global__ void k_zero(float* __restrict__ p, long n) {
  long i = (long)blockIdx.x * blockDim.x + threadIdx.x;
  if (i < n) p[i] = 0.f;
}

__global__ void k_cvt_f32_f16(const float* __restrict__ src, _Float16* __restrict__ dst, long n) {
  long i = (long)blockIdx.x * blockDim.x + threadIdx.x;
  if (i < n) dst[i] = (_Float16)src[i];
}

// node MLP: [N,4] -> lrelu([.,64]) -> lrelu([.,128]); tiny, scalar fp32
__global__ void k_node_mlp(const float* __restrict__ nf,
                           const float* __restrict__ w1, const float* __restrict__ b1,
                           const float* __restrict__ w2, const float* __restrict__ b2,
                           _Float16* __restrict__ out, int nrows) {
  int i = blockIdx.x * blockDim.x + threadIdx.x;
  if (i >= nrows) return;
  float x[4];
#pragma unroll
  for (int k = 0; k < 4; k++) x[k] = nf[(size_t)i * 4 + k];
  float h[64];
#pragma unroll
  for (int j = 0; j < 64; j++) {
    float s = b1[j];
#pragma unroll
    for (int k = 0; k < 4; k++) s += x[k] * w1[k * 64 + j];
    h[j] = lrelu(s);
  }
  for (int j = 0; j < HID; j++) {
    float s = b2[j];
#pragma unroll
    for (int k = 0; k < 64; k++) s += h[k] * w2[k * HID + j];
    out[(size_t)i * HID + j] = (_Float16)lrelu(s);
  }
}

// One wave = one 16-row tile. MODE 0: edge encoder (K1=32 padded from 3, A from fp32 feats,
// writes e fp32 + f16). MODE 1: e2 transform (K1=128, A from f16, in-place f16 out).
// MODE 2: edgeconv message (K1=256 = concat(n[dst], e), atomic scatter-sum to fp32 nodes).
template <int MODE, int KT1>
__global__ __launch_bounds__(256) void k_mlp16(
    const float* __restrict__ edge_feat,
    const _Float16* __restrict__ nodes_f16,
    const _Float16* __restrict__ edges_f16,
    const int* __restrict__ dst,
    const _Float16* __restrict__ w1t, const float* __restrict__ b1,
    const _Float16* __restrict__ w2t, const float* __restrict__ b2,
    float* __restrict__ out_f32,
    _Float16* __restrict__ out_f16) {
  __shared__ __align__(32) _Float16 h1buf[8][16 * 64];
  const int wid = threadIdx.x >> 5;
  const int lane = threadIdx.x & 31;
  const int g = lane >> 4;          // lane half-group (ISA A/B K grouping)
  const int col = lane & 15;        // N within tile (B/C/D)
  const int m = lane & 15;          // M row within tile (A)
  const int row0 = (blockIdx.x * 8 + wid) * 16;
  _Float16* h1 = &h1buf[wid][0];

  // ---------------- layer 1: [16,K1] x [K1,64] ----------------
  v8f c1[4];
#pragma unroll
  for (int j = 0; j < 4; j++) {
    const float bv = b1[j * 16 + col];
#pragma unroll
    for (int r = 0; r < 8; r++) c1[j][r] = bv;
  }
#pragma unroll
  for (int t = 0; t < KT1; t++) {
    v16h a;
    if constexpr (MODE == 0) {
#pragma unroll
      for (int i = 0; i < 16; i++) a[i] = (_Float16)0.f;
      if (g == 0) {  // only K=0..2 nonzero, held by low lane-group
        const float* x = edge_feat + (size_t)(row0 + m) * 3;
        a[0] = (_Float16)x[0]; a[1] = (_Float16)x[1]; a[2] = (_Float16)x[2];
      }
    } else if constexpr (MODE == 1) {
      const int f0 = t * 32 + 8 * g;
      v8h lo = *(const v8h*)(edges_f16 + (size_t)(row0 + m) * HID + f0);
      v8h hi = *(const v8h*)(edges_f16 + (size_t)(row0 + m) * HID + f0 + 16);
      a = packA(lo, hi);
    } else {
      const int d = dst[row0 + m];
      const int f0 = t * 32 + 8 * g;
      const int f1 = f0 + 16;
      v8h lo = (f0 < HID) ? *(const v8h*)(nodes_f16 + (size_t)d * HID + f0)
                          : *(const v8h*)(edges_f16 + (size_t)(row0 + m) * HID + (f0 - HID));
      v8h hi = (f1 < HID) ? *(const v8h*)(nodes_f16 + (size_t)d * HID + f1)
                          : *(const v8h*)(edges_f16 + (size_t)(row0 + m) * HID + (f1 - HID));
      a = packA(lo, hi);
    }
#pragma unroll
    for (int j = 0; j < 4; j++) {
      const _Float16* bp = w1t + (size_t)(j * 16 + col) * (KT1 * 32) + t * 32 + 8 * g;
      v16h bt = packA(*(const v8h*)bp, *(const v8h*)(bp + 16));
      c1[j] = __builtin_amdgcn_wmma_f32_16x16x32_f16(false, a, false, bt, (short)0, c1[j],
                                                     false, false);
    }
  }
  // lrelu + stage h1 in LDS as row-major [16][64] f16  (C layout: M = r + 8g, N = col)
#pragma unroll
  for (int j = 0; j < 4; j++) {
#pragma unroll
    for (int r = 0; r < 8; r++) {
      h1[(r + 8 * g) * 64 + j * 16 + col] = (_Float16)lrelu(c1[j][r]);
    }
  }
  __syncthreads();

  // ---------------- layer 2: [16,64] x [64,128] ----------------
  int drow[8];
  if constexpr (MODE == 2) {
#pragma unroll
    for (int r = 0; r < 8; r++) drow[r] = dst[row0 + r + 8 * g];
  }
#pragma unroll
  for (int j2 = 0; j2 < 8; j2++) {
    v8f c2;
    const float bv = b2[j2 * 16 + col];
#pragma unroll
    for (int r = 0; r < 8; r++) c2[r] = bv;
#pragma unroll
    for (int t2 = 0; t2 < 2; t2++) {
      v8h lo = *(const v8h*)(h1 + m * 64 + t2 * 32 + 8 * g);
      v8h hi = *(const v8h*)(h1 + m * 64 + t2 * 32 + 16 + 8 * g);
      v16h a2 = packA(lo, hi);
      const _Float16* bp2 = w2t + (size_t)(j2 * 16 + col) * 64 + t2 * 32 + 8 * g;
      v16h bt = packA(*(const v8h*)bp2, *(const v8h*)(bp2 + 16));
      c2 = __builtin_amdgcn_wmma_f32_16x16x32_f16(false, a2, false, bt, (short)0, c2,
                                                  false, false);
    }
#pragma unroll
    for (int r = 0; r < 8; r++) {
      const float v = lrelu(c2[r]);
      const int row = row0 + r + 8 * g;
      const int ca = j2 * 16 + col;
      if constexpr (MODE == 0) {
        out_f32[(size_t)row * HID + ca] = v;
        out_f16[(size_t)row * HID + ca] = (_Float16)v;
      } else if constexpr (MODE == 1) {
        out_f16[(size_t)row * HID + ca] = (_Float16)v;
      } else {
        atomicAdd(out_f32 + (size_t)drow[r] * HID + ca, v);
      }
    }
  }
}

// graph = n2.reshape(-1) @ graph_w : stream 655 MB of graph_w, coalesced 128-wide rows
__global__ __launch_bounds__(128) void k_graphmv(const float* __restrict__ n2,
                                                 const float* __restrict__ gw,
                                                 float* __restrict__ acc, int rows, int chunk) {
  const int t = threadIdx.x;
  const long r0 = (long)blockIdx.x * chunk;
  float a = 0.f;
  for (int i = 0; i < chunk; i++) {
    const long r = r0 + i;
    if (r < rows) {
      if (((i & 7) == 0) && (r + 16 < rows)) __builtin_prefetch(gw + (r + 16) * HID + t, 0, 1);
      a += n2[r] * gw[r * HID + t];
    }
  }
  atomicAdd(acc + t, a);
}

__global__ void k_final(const float* __restrict__ acc, const float* __restrict__ gb,
                        float* __restrict__ out) {
  int t = threadIdx.x;
  out[t] = lrelu(acc[t] + gb[t]);
}

extern "C" void kernel_launch(void* const* d_in, const int* in_sizes, int n_in,
                              void* d_out, int out_size, void* d_ws, size_t ws_size,
                              hipStream_t stream) {
  (void)in_sizes; (void)n_in; (void)out_size; (void)ws_size;
  const float* node_feat  = (const float*)d_in[0];
  const float* edge_feat  = (const float*)d_in[1];
  const int*   edge_index = (const int*)d_in[2];
  const float* node_w1 = (const float*)d_in[3];
  const float* node_b1 = (const float*)d_in[4];
  const float* node_w2 = (const float*)d_in[5];
  const float* node_b2 = (const float*)d_in[6];
  const float* edge_w1 = (const float*)d_in[7];
  const float* edge_b1 = (const float*)d_in[8];
  const float* edge_w2 = (const float*)d_in[9];
  const float* edge_b2 = (const float*)d_in[10];
  const float* g1_w1 = (const float*)d_in[11];
  const float* g1_b1 = (const float*)d_in[12];
  const float* g1_w2 = (const float*)d_in[13];
  const float* g1_b2 = (const float*)d_in[14];
  const float* g2_w1 = (const float*)d_in[15];
  const float* g2_b1 = (const float*)d_in[16];
  const float* g2_w2 = (const float*)d_in[17];
  const float* g2_b2 = (const float*)d_in[18];
  const float* g2e_w1 = (const float*)d_in[19];
  const float* g2e_b1 = (const float*)d_in[20];
  const float* g2e_w2 = (const float*)d_in[21];
  const float* g2e_b2 = (const float*)d_in[22];
  const float* graph_w = (const float*)d_in[23];
  const float* graph_b = (const float*)d_in[24];

  float* out_graph = (float*)d_out;
  float* out_e     = (float*)d_out + HID;

  char* ws = (char*)d_ws;
  size_t off = 0;
  auto take = [&](size_t bytes) -> char* {
    char* p = ws + off;
    off = (off + bytes + 255) & ~(size_t)255;
    return p;
  };
  _Float16* ef16   = (_Float16*)take((size_t)N_EDGES * HID * 2);  // e, then e2 in-place
  _Float16* nf16   = (_Float16*)take((size_t)N_NODES * HID * 2);  // n, then n1 in-place
  float*    n1f32  = (float*)take((size_t)N_NODES * HID * 4);
  float*    n2f32  = (float*)take((size_t)N_NODES * HID * 4);
  float*    acc    = (float*)take(HID * 4);
  _Float16* ew1t   = (_Float16*)take(64 * 32 * 2);
  _Float16* ew2t   = (_Float16*)take(128 * 64 * 2);
  _Float16* g1w1t  = (_Float16*)take(64 * 256 * 2);
  _Float16* g1w2t  = (_Float16*)take(128 * 64 * 2);
  _Float16* g2w1t  = (_Float16*)take(64 * 256 * 2);
  _Float16* g2w2t  = (_Float16*)take(128 * 64 * 2);
  _Float16* g2ew1t = (_Float16*)take(64 * 128 * 2);
  _Float16* g2ew2t = (_Float16*)take(128 * 64 * 2);

  auto tr = [&](_Float16* dstp, const float* srcp, int K, int Nc, int Kpad) {
    int total = Nc * Kpad;
    k_transpose_f16<<<(total + 255) / 256, 256, 0, stream>>>(dstp, srcp, K, Nc, Kpad);
  };
  tr(ew1t,   edge_w1, 3,   64,  32);
  tr(ew2t,   edge_w2, 64,  128, 64);
  tr(g1w1t,  g1_w1,   256, 64,  256);
  tr(g1w2t,  g1_w2,   64,  128, 64);
  tr(g2w1t,  g2_w1,   256, 64,  256);
  tr(g2w2t,  g2_w2,   64,  128, 64);
  tr(g2ew1t, g2e_w1,  128, 64,  128);
  tr(g2ew2t, g2e_w2,  64,  128, 64);

  {  // zero n1f32 .. acc (consecutive allocations, all sizes 256B-multiples)
    long nz = ((char*)acc + HID * 4 - (char*)n1f32) / 4;
    k_zero<<<(int)((nz + 255) / 256), 256, 0, stream>>>(n1f32, nz);
  }

  k_node_mlp<<<(N_NODES + 127) / 128, 128, 0, stream>>>(node_feat, node_w1, node_b1,
                                                        node_w2, node_b2, nf16, N_NODES);

  const int* dstp = edge_index + N_EDGES;  // edge_index[1]
  const int blocksE = N_EDGES / 16 / 8;    // 2500 blocks, 8 waves x 16 edges

  // e = mlp2(edge_feat)
  k_mlp16<0, 1><<<blocksE, 256, 0, stream>>>(edge_feat, nullptr, nullptr, nullptr,
                                             ew1t, edge_b1, ew2t, edge_b2, out_e, ef16);
  // n1 = scatter_sum(mlp2(concat(n[dst], e)))
  k_mlp16<2, 8><<<blocksE, 256, 0, stream>>>(nullptr, nf16, ef16, dstp,
                                             g1w1t, g1_b1, g1w2t, g1_b2, n1f32, nullptr);
  // e2 = mlp2(e)  (in-place over e_f16; conv1 already consumed e)
  k_mlp16<1, 4><<<blocksE, 256, 0, stream>>>(nullptr, nullptr, ef16, nullptr,
                                             g2ew1t, g2e_b1, g2ew2t, g2e_b2, nullptr, ef16);
  {  // n1 fp32 -> f16 (reuse nf16)
    long n = (long)N_NODES * HID;
    k_cvt_f32_f16<<<(int)((n + 255) / 256), 256, 0, stream>>>(n1f32, nf16, n);
  }
  // n2 = scatter_sum(mlp2(concat(n1[dst], e2)))
  k_mlp16<2, 8><<<blocksE, 256, 0, stream>>>(nullptr, nf16, ef16, dstp,
                                             g2w1t, g2_b1, g2w2t, g2_b2, n2f32, nullptr);

  const int rows = N_NODES * HID;  // 1,280,000
  const int chunk = 1000;
  k_graphmv<<<rows / chunk, 128, 0, stream>>>(n2f32, graph_w, acc, rows, chunk);
  k_final<<<1, 128, 0, stream>>>(acc, graph_b, out_graph);
}
